// TinyLlama_41248865911078
// MI455X (gfx1250) — matrix operand order repaired and checked by reading, not compiled, not run on hardware
//
#include <hip/hip_runtime.h>

// ---------------- model constants ----------------
#define LAYERS 4
#define DMODEL 2048
#define NHEAD  16
#define HD     128
#define SEQ    1024
#define BATCH  4
#define MROWS  (BATCH*SEQ)      // 4096
#define FFDIM  5120
#define VOCAB  32000

typedef __attribute__((ext_vector_type(16))) __bf16 v16bf;
typedef __attribute__((ext_vector_type(8)))  __bf16 v8bf;
typedef __attribute__((ext_vector_type(2)))  __bf16 v2bf;
typedef __attribute__((ext_vector_type(8)))  float  v8f;

typedef unsigned int u32x4 __attribute__((ext_vector_type(4)));
typedef int          i32x4 __attribute__((ext_vector_type(4)));
typedef int          i32x8 __attribute__((ext_vector_type(8)));

union BF16x16 { v16bf v; v8bf h[2]; };

#if defined(__has_builtin)
# if __has_builtin(__builtin_amdgcn_tensor_load_to_lds) && __has_builtin(__builtin_amdgcn_s_wait_tensorcnt)
#  define HAVE_TDM 1
# endif
#endif

__device__ __forceinline__ unsigned lds_off_u32(const void* p)
{
    return (unsigned)(unsigned long long)
        (__attribute__((address_space(3))) const char*)p;
}

#ifdef HAVE_TDM
// DMA a [128 rows x 32 cols] bf16 tile (row stride = kElems elements) from
// global to LDS, inserting 16B padding after every 64B row so the LDS image
// has an 80B (40 x bf16) pitch matching As[][40]. One call per wave0.
__device__ __forceinline__ void tdm_load_tile_bf16(
    const void* gptr, unsigned lds_byte_off, int kElems, int mRows)
{
    unsigned long long ga = (unsigned long long)gptr;
    unsigned lo  = (unsigned)__builtin_amdgcn_readfirstlane((int)(ga & 0xffffffffull));
    unsigned hi  = (unsigned)__builtin_amdgcn_readfirstlane((int)(ga >> 32));
    unsigned ldo = (unsigned)__builtin_amdgcn_readfirstlane((int)lds_byte_off);

    u32x4 g0;
    g0.x = 1u;                                   // count=1, user descriptor
    g0.y = ldo;                                  // lds_addr (bytes)
    g0.z = lo;                                   // global_addr[31:0]
    g0.w = (hi & 0x01ffffffu) | (2u << 30);      // global_addr[56:32] | type=2

    i32x8 g1;
    g1[0] = (1 << 16)       // data_size = 2 bytes
          | (1 << 20)       // pad_enable
          | (3 << 22)       // pad_interval: 16 DWORDs (=64B row)
          | (3 << 25);      // pad_amount:   4 DWORDs (=16B pad)
    g1[1] = (kElems & 0xffff) << 16;                               // tensor_dim0[15:0]
    g1[2] = ((kElems >> 16) & 0xffff) | ((mRows & 0xffff) << 16);  // dim0 hi | dim1 lo
    g1[3] = ((mRows  >> 16) & 0xffff) | (32 << 16);                // dim1 hi | tile_dim0=32
    g1[4] = 128;                                                   // tile_dim1=128 (tile_dim2=0)
    g1[5] = kElems;                                                // tensor_dim0_stride[31:0]
    g1[6] = 0;                                                     // stride hi | dim1_stride lo
    g1[7] = 0;

    i32x4 z4 = {0, 0, 0, 0};
#if __clang_major__ >= 23
    i32x8 z8 = {0, 0, 0, 0, 0, 0, 0, 0};
    __builtin_amdgcn_tensor_load_to_lds(g0, g1, z4, z4, z8, 0);
#else
    __builtin_amdgcn_tensor_load_to_lds(g0, g1, z4, z4, 0);
#endif
}
#endif

// =====================================================================
// GEMM: C[M,N] = A_bf16[M,K] (row-major) @ B_f32[K,N] (row-major)
// Tile 128x128x32, 256 threads = 8 waves, wave = 64x32 (4x2 wmma tiles).
// Double-buffered LDS pipeline; A tile staged via TDM (wave 0), B tile
// loaded to registers during compute and stored packed (b32) transposed.
// EPI: 0 = store f32, 1 = accumulate into f32 (residual add), 2 = store bf16
// =====================================================================
template<int EPI>
__global__ __launch_bounds__(256)
void gemm_kernel(const __bf16* __restrict__ A, const float* __restrict__ B,
                 float* __restrict__ Cf, __bf16* __restrict__ Cb,
                 int Mdim, int Ndim, int Kdim)
{
    __shared__ __align__(16) __bf16 As[2][128][40];   // [m][k], 80B pitch
    __shared__ __align__(16) __bf16 Bs[2][128][40];   // transposed [n][k]

    const int tid  = threadIdx.x;
    const int lane = tid & 31;
    const int w    = tid >> 5;
    const int wm   = w >> 2;        // 0..1 (64-row half)
    const int wn   = w & 3;         // 0..3 (32-col quarter)
    const int hi   = lane >> 4;     // lane half
    const int l16  = lane & 15;

    const int n0 = blockIdx.x * 128;
    const int m0 = blockIdx.y * 128;

    // B staging: thread -> k-pair (2 rows) x 8 consecutive n
    const int kp = tid >> 4;          // 0..15
    const int ng = (tid & 15) * 8;    // 0..120

    auto storeB = [&](int buf, float4 c0, float4 c1, float4 c2, float4 c3) {
        v2bf t;
        t[0]=(__bf16)c0.x; t[1]=(__bf16)c2.x; *(v2bf*)&Bs[buf][ng+0][2*kp]=t;
        t[0]=(__bf16)c0.y; t[1]=(__bf16)c2.y; *(v2bf*)&Bs[buf][ng+1][2*kp]=t;
        t[0]=(__bf16)c0.z; t[1]=(__bf16)c2.z; *(v2bf*)&Bs[buf][ng+2][2*kp]=t;
        t[0]=(__bf16)c0.w; t[1]=(__bf16)c2.w; *(v2bf*)&Bs[buf][ng+3][2*kp]=t;
        t[0]=(__bf16)c1.x; t[1]=(__bf16)c3.x; *(v2bf*)&Bs[buf][ng+4][2*kp]=t;
        t[0]=(__bf16)c1.y; t[1]=(__bf16)c3.y; *(v2bf*)&Bs[buf][ng+5][2*kp]=t;
        t[0]=(__bf16)c1.z; t[1]=(__bf16)c3.z; *(v2bf*)&Bs[buf][ng+6][2*kp]=t;
        t[0]=(__bf16)c1.w; t[1]=(__bf16)c3.w; *(v2bf*)&Bs[buf][ng+7][2*kp]=t;
    };

    v8f acc[4][2];
    #pragma unroll
    for (int i = 0; i < 4; i++)
        #pragma unroll
        for (int j = 0; j < 2; j++)
            #pragma unroll
            for (int r = 0; r < 8; r++) acc[i][j][r] = 0.0f;

    // ---- prologue: stage tile 0 ----
    {
        const float* B0 = B + n0;
        float4 c0 = *(const float4*)(B0 + (size_t)(2*kp)   * Ndim + ng);
        float4 c1 = *(const float4*)(B0 + (size_t)(2*kp)   * Ndim + ng + 4);
        float4 c2 = *(const float4*)(B0 + (size_t)(2*kp+1) * Ndim + ng);
        float4 c3 = *(const float4*)(B0 + (size_t)(2*kp+1) * Ndim + ng + 4);
        storeB(0, c0, c1, c2, c3);
#ifdef HAVE_TDM
        if (w == 0) {
            tdm_load_tile_bf16(A + (size_t)m0 * Kdim,
                               lds_off_u32(&As[0][0][0]), Kdim, Mdim);
            __builtin_amdgcn_s_wait_tensorcnt(0);
        }
#else
        #pragma unroll
        for (int i = 0; i < 2; i++) {
            int linear = tid + i * 256;
            int row = linear >> 2;
            int cg  = (linear & 3) * 8;
            *(v8bf*)(&As[0][row][cg]) =
                *(const v8bf*)(A + (size_t)(m0 + row) * Kdim + cg);
        }
#endif
        __syncthreads();
    }

    int p = 0;
    for (int k0 = 0; k0 < Kdim; k0 += 32) {
        const bool has_next = (k0 + 32) < Kdim;

        float4 nb0, nb1, nb2, nb3;
#ifndef HAVE_TDM
        v8bf na0, na1;
#endif
        if (has_next) {
            const float* Bn = B + (size_t)(k0 + 32) * Ndim + n0;
            nb0 = *(const float4*)(Bn + (size_t)(2*kp)   * Ndim + ng);
            nb1 = *(const float4*)(Bn + (size_t)(2*kp)   * Ndim + ng + 4);
            nb2 = *(const float4*)(Bn + (size_t)(2*kp+1) * Ndim + ng);
            nb3 = *(const float4*)(Bn + (size_t)(2*kp+1) * Ndim + ng + 4);
#ifdef HAVE_TDM
            if (w == 0)
                tdm_load_tile_bf16(A + (size_t)m0 * Kdim + (k0 + 32),
                                   lds_off_u32(&As[p ^ 1][0][0]), Kdim, Mdim);
#else
            {
                int l0 = tid, l1 = tid + 256;
                na0 = *(const v8bf*)(A + (size_t)(m0 + (l0 >> 2)) * Kdim + (k0+32) + (l0 & 3) * 8);
                na1 = *(const v8bf*)(A + (size_t)(m0 + (l1 >> 2)) * Kdim + (k0+32) + (l1 & 3) * 8);
            }
#endif
        }

        // ---- fragments + WMMA from buffer p ----
        BF16x16 af[4], bf[2];
        #pragma unroll
        for (int i = 0; i < 4; i++) {
            int m = wm * 64 + i * 16 + l16;
            af[i].h[0] = *(const v8bf*)&As[p][m][hi * 8];
            af[i].h[1] = *(const v8bf*)&As[p][m][16 + hi * 8];
        }
        #pragma unroll
        for (int j = 0; j < 2; j++) {
            int n = wn * 32 + j * 16 + l16;
            bf[j].h[0] = *(const v8bf*)&Bs[p][n][hi * 8];
            bf[j].h[1] = *(const v8bf*)&Bs[p][n][16 + hi * 8];
        }
        #pragma unroll
        for (int i = 0; i < 4; i++)
            #pragma unroll
            for (int j = 0; j < 2; j++)
                acc[i][j] = __builtin_amdgcn_wmma_f32_16x16x32_bf16(
                    false, af[i].v, false, bf[j].v, (short)0, acc[i][j], false, false);

        if (has_next) {
            storeB(p ^ 1, nb0, nb1, nb2, nb3);
#ifndef HAVE_TDM
            {
                int l0 = tid, l1 = tid + 256;
                *(v8bf*)(&As[p ^ 1][l0 >> 2][(l0 & 3) * 8]) = na0;
                *(v8bf*)(&As[p ^ 1][l1 >> 2][(l1 & 3) * 8]) = na1;
            }
#endif
        }
#ifdef HAVE_TDM
        if (w == 0) __builtin_amdgcn_s_wait_tensorcnt(0);
#endif
        __syncthreads();
        p ^= 1;
    }

    // ---- epilogue (C layout: vgpr r = row, lane = col, hi half = row+8) ----
    #pragma unroll
    for (int i = 0; i < 4; i++)
        #pragma unroll
        for (int j = 0; j < 2; j++)
            #pragma unroll
            for (int r = 0; r < 8; r++) {
                int row = m0 + wm * 64 + i * 16 + hi * 8 + r;
                int col = n0 + wn * 32 + j * 16 + l16;
                size_t idx = (size_t)row * Ndim + col;
                if (EPI == 0)      Cf[idx] = acc[i][j][r];
                else if (EPI == 1) Cf[idx] += acc[i][j][r];
                else               Cb[idx] = (__bf16)acc[i][j][r];
            }
}

// =====================================================================
// Flash attention: one block = (batch b, head h, 64-row q tile).
// 8 waves; scores via WMMA (Q@K^T), online softmax, P@V via WMMA.
// =====================================================================
__global__ __launch_bounds__(256)
void attn_kernel(const __bf16* __restrict__ Q, const __bf16* __restrict__ K,
                 const __bf16* __restrict__ Vv, __bf16* __restrict__ O)
{
    __shared__ __align__(16) __bf16 Qs[64][136];
    __shared__ __align__(16) __bf16 Ks[64][136];
    __shared__ __align__(16) __bf16 Vt[128][72];   // transposed [hd][key]
    __shared__ __align__(16) __bf16 Ps[64][72];    // probs [q][key]
    __shared__ float row_m[64], row_l[64], row_scale[64];
    __shared__ float red_max[64][2], red_sum[64][2];

    const int tid  = threadIdx.x;
    const int lane = tid & 31;
    const int w    = tid >> 5;
    const int hi   = lane >> 4;
    const int l16  = lane & 15;

    const int qt   = blockIdx.x;          // 0..15
    const int head = blockIdx.y;
    const int b    = blockIdx.z;
    const int q0   = qt * 64;

    const __bf16* qbase = Q + ((size_t)b * SEQ + q0) * DMODEL + head * HD;

    // stage Q tile (64x128 bf16)
    #pragma unroll
    for (int i = 0; i < 4; i++) {
        int linear = tid + i * 256;
        int r = linear >> 4;
        int cg = (linear & 15) * 8;
        *(v8bf*)(&Qs[r][cg]) = *(const v8bf*)(qbase + (size_t)r * DMODEL + cg);
    }
    if (tid < 64) { row_m[tid] = -3.0e38f; row_l[tid] = 0.0f; }

    const int om = w >> 1;     // 16-row O strip
    const int on = w & 1;      // 64-col hd half
    v8f oacc[4];
    #pragma unroll
    for (int t = 0; t < 4; t++)
        #pragma unroll
        for (int r = 0; r < 8; r++) oacc[t][r] = 0.0f;

    const float scale = 0.08838834764831845f;   // 1/sqrt(128)

    for (int kt = 0; kt <= qt; kt++) {
        const __bf16* kbase = K  + ((size_t)b * SEQ + kt * 64) * DMODEL + head * HD;
        const __bf16* vbase = Vv + ((size_t)b * SEQ + kt * 64) * DMODEL + head * HD;
        #pragma unroll
        for (int i = 0; i < 4; i++) {
            int linear = tid + i * 256;
            int r = linear >> 4;
            int cg = (linear & 15) * 8;
            *(v8bf*)(&Ks[r][cg]) = *(const v8bf*)(kbase + (size_t)r * DMODEL + cg);
            v8bf vv = *(const v8bf*)(vbase + (size_t)r * DMODEL + cg);
            #pragma unroll
            for (int e = 0; e < 8; e++) Vt[cg + e][r] = vv[e];
        }
        if (tid < 64) { red_max[tid][0] = -3.0e38f; red_max[tid][1] = -3.0e38f; }
        __syncthreads();

        // ---- scores: wave -> (16 q rows, 32 keys) ----
        const int sm = w >> 1, sn = w & 1;
        v8f sc[2];
        #pragma unroll
        for (int j = 0; j < 2; j++)
            #pragma unroll
            for (int r = 0; r < 8; r++) sc[j][r] = 0.0f;

        #pragma unroll
        for (int ko = 0; ko < 128; ko += 32) {
            BF16x16 af;
            int mrow = sm * 16 + l16;
            af.h[0] = *(const v8bf*)&Qs[mrow][ko + hi * 8];
            af.h[1] = *(const v8bf*)&Qs[mrow][ko + 16 + hi * 8];
            #pragma unroll
            for (int j = 0; j < 2; j++) {
                BF16x16 bfr;
                int krow = (sn * 2 + j) * 16 + l16;
                bfr.h[0] = *(const v8bf*)&Ks[krow][ko + hi * 8];
                bfr.h[1] = *(const v8bf*)&Ks[krow][ko + 16 + hi * 8];
                sc[j] = __builtin_amdgcn_wmma_f32_16x16x32_bf16(
                    false, af.v, false, bfr.v, (short)0, sc[j], false, false);
            }
        }
        // scale + causal mask + per-row max partials
        int qrow_base = q0 + sm * 16 + hi * 8;
        int kcol0 = kt * 64 + (sn * 2) * 16 + l16;
        #pragma unroll
        for (int r = 0; r < 8; r++) {
            int qrow = qrow_base + r;
            float v0 = sc[0][r] * scale; if (kcol0      > qrow) v0 = -3.0e38f;
            float v1 = sc[1][r] * scale; if (kcol0 + 16 > qrow) v1 = -3.0e38f;
            sc[0][r] = v0; sc[1][r] = v1;
            float vm = fmaxf(v0, v1);
            vm = fmaxf(vm, __shfl_xor(vm, 1, 16));
            vm = fmaxf(vm, __shfl_xor(vm, 2, 16));
            vm = fmaxf(vm, __shfl_xor(vm, 4, 16));
            vm = fmaxf(vm, __shfl_xor(vm, 8, 16));
            if (l16 == 0) red_max[sm * 16 + hi * 8 + r][sn] = vm;
        }
        __syncthreads();
        if (tid < 64) {
            float nm = fmaxf(row_m[tid], fmaxf(red_max[tid][0], red_max[tid][1]));
            row_scale[tid] = __expf(row_m[tid] - nm);
            row_m[tid] = nm;
            red_sum[tid][0] = 0.0f; red_sum[tid][1] = 0.0f;
        }
        __syncthreads();
        // exp -> Ps (bf16), row-sum partials
        #pragma unroll
        for (int r = 0; r < 8; r++) {
            int lrow = sm * 16 + hi * 8 + r;
            float nm = row_m[lrow];
            float p0 = __expf(sc[0][r] - nm);
            float p1 = __expf(sc[1][r] - nm);
            Ps[lrow][(sn * 2) * 16 + l16]     = (__bf16)p0;
            Ps[lrow][(sn * 2 + 1) * 16 + l16] = (__bf16)p1;
            float ps = p0 + p1;
            ps += __shfl_xor(ps, 1, 16);
            ps += __shfl_xor(ps, 2, 16);
            ps += __shfl_xor(ps, 4, 16);
            ps += __shfl_xor(ps, 8, 16);
            if (l16 == 0) red_sum[lrow][sn] = ps;
        }
        __syncthreads();
        if (tid < 64)
            row_l[tid] = row_l[tid] * row_scale[tid] + red_sum[tid][0] + red_sum[tid][1];

        // ---- rescale O accumulators, then P@V ----
        #pragma unroll
        for (int t = 0; t < 4; t++)
            #pragma unroll
            for (int r = 0; r < 8; r++)
                oacc[t][r] *= row_scale[om * 16 + hi * 8 + r];

        #pragma unroll
        for (int ko = 0; ko < 64; ko += 32) {
            BF16x16 af;
            int mrow = om * 16 + l16;
            af.h[0] = *(const v8bf*)&Ps[mrow][ko + hi * 8];
            af.h[1] = *(const v8bf*)&Ps[mrow][ko + 16 + hi * 8];
            #pragma unroll
            for (int t = 0; t < 4; t++) {
                BF16x16 bfr;
                int nrow = on * 64 + t * 16 + l16;
                bfr.h[0] = *(const v8bf*)&Vt[nrow][ko + hi * 8];
                bfr.h[1] = *(const v8bf*)&Vt[nrow][ko + 16 + hi * 8];
                oacc[t] = __builtin_amdgcn_wmma_f32_16x16x32_bf16(
                    false, af.v, false, bfr.v, (short)0, oacc[t], false, false);
            }
        }
        __syncthreads();
    }

    // normalize + store bf16
    __bf16* obase = O + ((size_t)b * SEQ + q0) * DMODEL + head * HD;
    #pragma unroll
    for (int t = 0; t < 4; t++)
        #pragma unroll
        for (int r = 0; r < 8; r++) {
            int lrow = om * 16 + hi * 8 + r;
            float val = oacc[t][r] / row_l[lrow];
            int col = on * 64 + t * 16 + l16;
            obase[(size_t)lrow * DMODEL + col] = (__bf16)val;
        }
}

// =====================================================================
// helpers
// =====================================================================
__global__ __launch_bounds__(256)
void embed_kernel(const int* __restrict__ tok, const float* __restrict__ emb,
                  float* __restrict__ h)
{
    int row = blockIdx.x;
    int t = tok[row];
    const float* src = emb + (size_t)t * DMODEL;
    float* dst = h + (size_t)row * DMODEL;
    #pragma unroll
    for (int i = 0; i < 2; i++) {
        int c = threadIdx.x + i * 256;
        *(float4*)(dst + c * 4) = *(const float4*)(src + c * 4);
    }
}

__global__ __launch_bounds__(256)
void rmsnorm_kernel(const float* __restrict__ X, const float* __restrict__ Wt,
                    __bf16* __restrict__ Yb)
{
    __shared__ float red[8];
    const int row = blockIdx.x;
    const float* x = X + (size_t)row * DMODEL;
    float4 vv[2];
    float ss = 0.0f;
    #pragma unroll
    for (int i = 0; i < 2; i++) {
        int c = threadIdx.x + i * 256;
        float4 v = *(const float4*)(x + c * 4);
        vv[i] = v;
        ss += v.x * v.x + v.y * v.y + v.z * v.z + v.w * v.w;
    }
    #pragma unroll
    for (int off = 16; off > 0; off >>= 1) ss += __shfl_down(ss, off, 32);
    if ((threadIdx.x & 31) == 0) red[threadIdx.x >> 5] = ss;
    __syncthreads();
    if (threadIdx.x == 0) {
        float t = 0.0f;
        #pragma unroll
        for (int i = 0; i < 8; i++) t += red[i];
        red[0] = rsqrtf(t / (float)DMODEL + 1e-5f);
    }
    __syncthreads();
    float inv = red[0];
    #pragma unroll
    for (int i = 0; i < 2; i++) {
        int c = threadIdx.x + i * 256;
        float4 v = vv[i];
        float4 wv = *(const float4*)(Wt + c * 4);
        __bf16* o = Yb + (size_t)row * DMODEL + c * 4;
        o[0] = (__bf16)(v.x * inv * wv.x);
        o[1] = (__bf16)(v.y * inv * wv.y);
        o[2] = (__bf16)(v.z * inv * wv.z);
        o[3] = (__bf16)(v.w * inv * wv.w);
    }
}

__global__ __launch_bounds__(256)
void rope_kernel(const float* __restrict__ X, __bf16* __restrict__ Yb,
                 const int* __restrict__ start_pos)
{
    const int row = blockIdx.x;              // b*SEQ + s
    const int s = row & (SEQ - 1);
    const float pos = (float)(s + start_pos[0]);
    const float* x = X + (size_t)row * DMODEL;
    __bf16* y = Yb + (size_t)row * DMODEL;
    #pragma unroll
    for (int i = 0; i < 4; i++) {
        int p  = threadIdx.x + i * 256;      // pair index 0..1023
        int pj = p & 63;                     // pair within head (hd/2 = 64)
        float inv = __expf((float)(2 * pj) * (-9.210340371976184f / 128.0f));
        float f = pos * inv;
        float c = __cosf(f), sn = __sinf(f);
        int base = (p >> 6) * HD + pj * 2;
        float a = x[base], bb = x[base + 1];
        y[base]     = (__bf16)(a * c - bb * sn);
        y[base + 1] = (__bf16)(a * sn + bb * c);
    }
}

__global__ __launch_bounds__(256)
void silu_mul_kernel(const float* __restrict__ G, const float* __restrict__ U,
                     __bf16* __restrict__ A2)
{
    size_t c = (size_t)blockIdx.x * 256 + threadIdx.x;
    float4 g = *(const float4*)(G + c * 4);
    float4 u = *(const float4*)(U + c * 4);
    __bf16* o = A2 + c * 4;
    o[0] = (__bf16)(g.x / (1.0f + __expf(-g.x)) * u.x);
    o[1] = (__bf16)(g.y / (1.0f + __expf(-g.y)) * u.y);
    o[2] = (__bf16)(g.z / (1.0f + __expf(-g.z)) * u.z);
    o[3] = (__bf16)(g.w / (1.0f + __expf(-g.w)) * u.w);
}

// =====================================================================
extern "C" void kernel_launch(void* const* d_in, const int* in_sizes, int n_in,
                              void* d_out, int out_size, void* d_ws, size_t ws_size,
                              hipStream_t stream)
{
    (void)in_sizes; (void)n_in; (void)out_size; (void)ws_size;
    const int*   tok    = (const int*)  d_in[0];
    const float* emb    = (const float*)d_in[1];
    const float* wq     = (const float*)d_in[2];
    const float* wk     = (const float*)d_in[3];
    const float* wv     = (const float*)d_in[4];
    const float* wo     = (const float*)d_in[5];
    const float* wg     = (const float*)d_in[6];
    const float* wu     = (const float*)d_in[7];
    const float* wd     = (const float*)d_in[8];
    const float* in_n   = (const float*)d_in[9];
    const float* post_n = (const float*)d_in[10];
    const float* out_n  = (const float*)d_in[11];
    const float* wout   = (const float*)d_in[12];
    const int*   spos   = (const int*)  d_in[13];
    float* out = (float*)d_out;

    char* wsp = (char*)d_ws;
    auto alloc = [&](size_t bytes) {
        char* p = wsp;
        wsp += (bytes + 255) & ~(size_t)255;
        return p;
    };
    float*  h    = (float*) alloc((size_t)MROWS * DMODEL * 4);
    __bf16* xn   = (__bf16*)alloc((size_t)MROWS * DMODEL * 2);
    float*  gbuf = (float*) alloc((size_t)MROWS * FFDIM * 4);   // aliased as qf
    float*  ubuf = (float*) alloc((size_t)MROWS * FFDIM * 4);   // aliased as kf
    __bf16* act  = (__bf16*)alloc((size_t)MROWS * FFDIM * 2);
    __bf16* qb   = (__bf16*)alloc((size_t)MROWS * DMODEL * 2);
    __bf16* kb   = (__bf16*)alloc((size_t)MROWS * DMODEL * 2);
    __bf16* vb   = (__bf16*)alloc((size_t)MROWS * DMODEL * 2);
    __bf16* ob   = (__bf16*)alloc((size_t)MROWS * DMODEL * 2);
    float*  qf   = gbuf;   // f32 q/k temps reuse FFN buffers (disjoint lifetimes)
    float*  kf   = ubuf;

    auto gemm_f32 = [&](const __bf16* A, const float* B, float* C, int Mm, int Nn, int Kk) {
        dim3 g(Nn / 128, Mm / 128);
        gemm_kernel<0><<<g, 256, 0, stream>>>(A, B, C, nullptr, Mm, Nn, Kk);
    };
    auto gemm_acc = [&](const __bf16* A, const float* B, float* C, int Mm, int Nn, int Kk) {
        dim3 g(Nn / 128, Mm / 128);
        gemm_kernel<1><<<g, 256, 0, stream>>>(A, B, C, nullptr, Mm, Nn, Kk);
    };
    auto gemm_bf = [&](const __bf16* A, const float* B, __bf16* C, int Mm, int Nn, int Kk) {
        dim3 g(Nn / 128, Mm / 128);
        gemm_kernel<2><<<g, 256, 0, stream>>>(A, B, nullptr, C, Mm, Nn, Kk);
    };

    embed_kernel<<<MROWS, 256, 0, stream>>>(tok, emb, h);

    for (int l = 0; l < LAYERS; l++) {
        const float* wq_l = wq + (size_t)l * DMODEL * DMODEL;
        const float* wk_l = wk + (size_t)l * DMODEL * DMODEL;
        const float* wv_l = wv + (size_t)l * DMODEL * DMODEL;
        const float* wo_l = wo + (size_t)l * DMODEL * DMODEL;
        const float* wg_l = wg + (size_t)l * DMODEL * FFDIM;
        const float* wu_l = wu + (size_t)l * DMODEL * FFDIM;
        const float* wd_l = wd + (size_t)l * FFDIM * DMODEL;

        rmsnorm_kernel<<<MROWS, 256, 0, stream>>>(h, in_n + (size_t)l * DMODEL, xn);

        gemm_f32(xn, wq_l, qf, MROWS, DMODEL, DMODEL);
        gemm_f32(xn, wk_l, kf, MROWS, DMODEL, DMODEL);
        gemm_bf (xn, wv_l, vb, MROWS, DMODEL, DMODEL);

        rope_kernel<<<MROWS, 256, 0, stream>>>(qf, qb, spos);
        rope_kernel<<<MROWS, 256, 0, stream>>>(kf, kb, spos);

        attn_kernel<<<dim3(SEQ / 64, NHEAD, BATCH), 256, 0, stream>>>(qb, kb, vb, ob);

        gemm_acc(ob, wo_l, h, MROWS, DMODEL, DMODEL);

        rmsnorm_kernel<<<MROWS, 256, 0, stream>>>(h, post_n + (size_t)l * DMODEL, xn);

        gemm_f32(xn, wg_l, gbuf, MROWS, FFDIM, DMODEL);
        gemm_f32(xn, wu_l, ubuf, MROWS, FFDIM, DMODEL);

        silu_mul_kernel<<<((size_t)MROWS * FFDIM) / 1024, 256, 0, stream>>>(gbuf, ubuf, act);

        gemm_acc(act, wd_l, h, MROWS, DMODEL, FFDIM);
    }

    rmsnorm_kernel<<<MROWS, 256, 0, stream>>>(h, out_n, xn);
    gemm_f32(xn, wout, out, MROWS, VOCAB, DMODEL);
}
